// differentiableSuperpixelTokenizer_63204738728311
// MI455X (gfx1250) — matrix operand order repaired and checked by reading, not compiled, not run on hardware
//
#include <hip/hip_runtime.h>
#include <hip/hip_bf16.h>
#include <math.h>

typedef _Float16 f16;
typedef __attribute__((ext_vector_type(16))) _Float16 v16h;
typedef __attribute__((ext_vector_type(8)))  _Float16 v8h;
typedef __attribute__((ext_vector_type(8)))  float    v8f;

#define BATCH 8
#define HIN   224
#define C1    64
#define HF    112
#define SP    (HF*HF)        /* 12544 */
#define C2    768
#define HIDN  512
#define NSEG  196
#define ROWS  (BATCH*SP)     /* 100352 */
#define KC2   576            /* conv2 GEMM K = 64*9 */
#define BNEPS 1e-5f
#define CNT_F 100352.0f      /* BN stat count = B*HF*HF */

// ------------------------------------------------------------------
// CDNA5 async global->LDS staging (guarded; falls back to ds_store)
// Builtin signature (from compiler diagnostic): b128 variant takes
//   (v4i __device__(AS1)*, v4i __shared__(AS3)*, imm offset, imm cpol)
// ------------------------------------------------------------------
#if defined(__has_builtin)
#if __has_builtin(__builtin_amdgcn_global_load_async_to_lds_b128)
#define HAS_ASYNC_LDS 1
#endif
#endif

typedef int v4i __attribute__((vector_size(16)));
typedef __attribute__((address_space(1))) v4i gv4i;
typedef __attribute__((address_space(3))) v4i lv4i;

__device__ __forceinline__ void stage16(const f16* __restrict__ g, f16* __restrict__ l) {
#ifdef HAS_ASYNC_LDS
  __builtin_amdgcn_global_load_async_to_lds_b128((gv4i*)(uintptr_t)g, (lv4i*)l, 0, 0);
#else
  *(v8h*)l = *(const v8h*)g;
#endif
}
__device__ __forceinline__ void stage_join() {
#ifdef HAS_ASYNC_LDS
  __builtin_amdgcn_s_wait_asynccnt(0);
#endif
}

// fragment helpers: ISA 16-bit A 16x32 layout => two contiguous 8-half chunks
__device__ __forceinline__ v16h frag_from_chunks(v8h lo, v8h hi) {
  v16h a;
  #pragma unroll
  for (int i = 0; i < 8; ++i) { a[i] = lo[i]; a[i+8] = hi[i]; }
  return a;
}

// ---------------------------------------------------------------- utilities
__global__ void k_zero_f32(float* p, int n) {
  for (int i = blockIdx.x*256 + threadIdx.x; i < n; i += gridDim.x*256) p[i] = 0.0f;
}
__global__ void k_fill_i32(int* p, int n, int v) {
  for (int i = blockIdx.x*256 + threadIdx.x; i < n; i += gridDim.x*256) p[i] = v;
}

// ---------------------------------------------------------------- conv1 (naive, ~1% of FLOPs)
__global__ void k_conv1(const float* __restrict__ img, const float* __restrict__ w,
                        const float* __restrict__ bias, float* __restrict__ out) {
  int idx = blockIdx.x*256 + threadIdx.x;
  const int total = BATCH*C1*SP;
  if (idx >= total) return;
  int ow = idx % HF; int t = idx / HF;
  int oh = t % HF;   t /= HF;
  int oc = t % C1;   int b = t / C1;
  float acc = bias[oc];
  for (int ic = 0; ic < 3; ++ic) {
    const float* ip = img + ((size_t)(b*3 + ic))*HIN*HIN;
    const float* wp = w + ((size_t)(oc*3 + ic))*49;
    #pragma unroll
    for (int kh = 0; kh < 7; ++kh) {
      int y = oh*2 + kh - 3;
      if ((unsigned)y >= HIN) continue;
      #pragma unroll
      for (int kw = 0; kw < 7; ++kw) {
        int x = ow*2 + kw - 3;
        if ((unsigned)x >= HIN) continue;
        acc = fmaf(ip[y*HIN + x], wp[kh*7 + kw], acc);
      }
    }
  }
  out[idx] = acc;
}

// ---------------------------------------------------------------- BN stats (per-channel sum/sumsq)
__global__ void k_stats_f32(const float* __restrict__ x, float* __restrict__ stats, int C) {
  int c = blockIdx.x;
  __shared__ float ssum[256], ssq[256];
  float s = 0.f, q = 0.f;
  for (int i = blockIdx.y*256 + threadIdx.x; i < BATCH*SP; i += gridDim.y*256) {
    int b = i / SP, sp = i % SP;
    float v = x[((size_t)(b*C + c))*SP + sp];
    s += v; q += v*v;
  }
  ssum[threadIdx.x] = s; ssq[threadIdx.x] = q;
  __syncthreads();
  for (int wd = 128; wd > 0; wd >>= 1) {
    if (threadIdx.x < wd) { ssum[threadIdx.x] += ssum[threadIdx.x+wd]; ssq[threadIdx.x] += ssq[threadIdx.x+wd]; }
    __syncthreads();
  }
  if (threadIdx.x == 0) { atomicAdd(&stats[2*c], ssum[0]); atomicAdd(&stats[2*c+1], ssq[0]); }
}

__global__ void k_stats_f16(const f16* __restrict__ x, float* __restrict__ stats, int C) {
  int c = blockIdx.x;
  __shared__ float ssum[256], ssq[256];
  float s = 0.f, q = 0.f;
  for (int i = blockIdx.y*256 + threadIdx.x; i < BATCH*SP; i += gridDim.y*256) {
    int b = i / SP, sp = i % SP;
    float v = (float)x[((size_t)(b*C + c))*SP + sp];
    s += v; q += v*v;
  }
  ssum[threadIdx.x] = s; ssq[threadIdx.x] = q;
  __syncthreads();
  for (int wd = 128; wd > 0; wd >>= 1) {
    if (threadIdx.x < wd) { ssum[threadIdx.x] += ssum[threadIdx.x+wd]; ssq[threadIdx.x] += ssq[threadIdx.x+wd]; }
    __syncthreads();
  }
  if (threadIdx.x == 0) { atomicAdd(&stats[2*c], ssum[0]); atomicAdd(&stats[2*c+1], ssq[0]); }
}

// ---------------------------------------------------------------- BN + ReLU apply
__global__ void k_bn_relu_1(const float* __restrict__ x, const float* __restrict__ stats,
                            const float* __restrict__ g, const float* __restrict__ bb,
                            f16* __restrict__ out) {
  int idx = blockIdx.x*256 + threadIdx.x;
  const int total = BATCH*C1*SP;
  if (idx >= total) return;
  int c = (idx / SP) % C1;
  float mu = stats[2*c] * (1.0f/CNT_F);
  float var = stats[2*c+1] * (1.0f/CNT_F) - mu*mu;
  float rs = rsqrtf(var + BNEPS);
  float v = (x[idx] - mu) * rs * g[c] + bb[c];
  out[idx] = (f16)fmaxf(v, 0.0f);
}

__global__ void k_bn_relu_2(f16* __restrict__ x, const float* __restrict__ stats,
                            const float* __restrict__ g, const float* __restrict__ bb) {
  int idx = blockIdx.x*256 + threadIdx.x;
  const int total = BATCH*C2*SP;
  if (idx >= total) return;
  int c = (idx / SP) % C2;
  float mu = stats[2*c] * (1.0f/CNT_F);
  float var = stats[2*c+1] * (1.0f/CNT_F) - mu*mu;
  float rs = rsqrtf(var + BNEPS);
  float v = ((float)x[idx] - mu) * rs * g[c] + bb[c];
  x[idx] = (f16)fmaxf(v, 0.0f);
}

// ---------------------------------------------------------------- weight prep
// wA  [768][576]  k = tap*64+ic (conv2 A, row-major)
// w1t [512][768]  = W1^T   (B fragments become contiguous)
// w2t [768][512]  = W2^T
__global__ void k_prep_weights(const float* __restrict__ conv2w, const float* __restrict__ W1,
                               const float* __restrict__ W2, f16* __restrict__ wA,
                               f16* __restrict__ w1t, f16* __restrict__ w2t) {
  int i = blockIdx.x*256 + threadIdx.x;
  if (i < C2*KC2) {
    int oc = i / KC2, k = i % KC2;
    int tap = k / C1, ic = k % C1;
    wA[i] = (f16)conv2w[(size_t)oc*KC2 + ic*9 + tap];
  }
  if (i < HIDN*C2) {                       // w1t[n][k] = W1[k][n]
    int n = i / C2, k = i % C2;
    w1t[i] = (f16)W1[(size_t)k*HIDN + n];
  }
  if (i < C2*HIDN) {                       // w2t[n][k] = W2[k][n]
    int n = i / HIDN, k = i % HIDN;
    w2t[i] = (f16)W2[(size_t)k*C2 + n];
  }
}

// ---------------------------------------------------------------- conv2 as implicit GEMM, WMMA f16->f32
// M=768 (oc), N=100352 (b,h,w), K=576. One wave -> 4 oc-tiles x 16 positions.
// K-loop fully unrolled: tap geometry becomes compile-time constants.
#define OCPW 4   /* oc tiles per wave */
__global__ void k_conv2_wmma(const f16* __restrict__ f1, const f16* __restrict__ wA,
                             const float* __restrict__ bias, f16* __restrict__ out) {
  int ocg = blockIdx.x;               // 0..11 -> oc tiles [ocg*4, ocg*4+4)
  int t = blockIdx.y;                 // 0..6271  (b*HF*7)
  int wt = t % 7; t /= 7;
  int h  = t % HF;
  int b  = t / HF;
  int w0 = wt*16;
  int lane = threadIdx.x;
  int half = lane >> 4;
  int lm   = lane & 15;
  const f16* ap0 = wA + (size_t)(ocg*OCPW*16 + lm)*KC2;
  v8f acc[OCPW] = {};
  #pragma unroll
  for (int ks = 0; ks < 18; ++ks) {
    const int tap = ks >> 1;
    const int kh = tap / 3, kw = tap % 3;
    const int icb = (ks & 1) * 32;
    // ---- B gather (padded), shared by all 4 oc tiles
    v16h bf;
    {
      int y = h + kh - 1;
      int x = w0 + lm + kw - 1;
      bool ok = ((unsigned)y < HF) && ((unsigned)x < HF);
      #pragma unroll
      for (int v = 0; v < 8; ++v) {
        int ic0 = icb + (half << 3) + 2*v + ((v >> 2) << 3); // B 32x16: table order, half h -> K 8h..8h+7, 16+8h..16+8h+7
        f16 v0 = (f16)0, v1 = (f16)0;
        if (ok) {
          v0 = f1[((size_t)(b*C1 + ic0    )*HF + y)*HF + x];
          v1 = f1[((size_t)(b*C1 + ic0 + 1)*HF + y)*HF + x];
        }
        bf[2*v] = v0; bf[2*v+1] = v1;
      }
    }
    if (ks + 2 < 18) __builtin_prefetch(ap0 + (ks+2)*32, 0, 3);
    // ---- 4 A fragments (contiguous 16B chunks) + 4 WMMAs
    #pragma unroll
    for (int o = 0; o < OCPW; ++o) {
      const f16* ap = ap0 + (size_t)o*16*KC2 + ks*32;
      v8h lo = *(const v8h*)(ap + half*8);
      v8h hi = *(const v8h*)(ap + 16 + half*8);
      v16h a = frag_from_chunks(lo, hi);
      acc[o] = __builtin_amdgcn_wmma_f32_16x16x32_f16(false, a, false, bf, (short)0, acc[o], false, false);
    }
  }
  int x = w0 + lm;
  #pragma unroll
  for (int o = 0; o < OCPW; ++o) {
    #pragma unroll
    for (int g = 0; g < 8; ++g) {
      int oc = (ocg*OCPW + o)*16 + g + 8*half;            // C/D layout: lane half -> M+8
      float v = acc[o][g] + bias[oc];
      out[((size_t)(b*C2 + oc)*HF + h)*HF + x] = (f16)v;
    }
  }
}

// ---------------------------------------------------------------- segment downsample + last-write-wins winner
__global__ void k_seg(const int* __restrict__ seg, int* __restrict__ seg_flat,
                      int* __restrict__ winner) {
  int r = blockIdx.x*256 + threadIdx.x;
  if (r >= ROWS) return;
  int b = r / SP, n = r % SP;
  int h = n / HF, w = n % HF;
  int s = seg[((size_t)b*HIN + h*2)*HIN + w*2] % NSEG;     // nearest: src = 2*dst
  seg_flat[r] = s;
  atomicMax(&winner[b*NSEG + s], r);
}

// ---------------------------------------------------------------- fold pos-embed into f2 in place: x = feats + pos
__global__ void k_addpos(f16* __restrict__ f2, const float* __restrict__ pos,
                         const int* __restrict__ seg_flat) {
  int idx = blockIdx.x*256 + threadIdx.x;
  const int total = ROWS*C2;
  if (idx >= total) return;
  int r = idx / C2, c = idx % C2;
  float v = (float)f2[idx] + pos[(size_t)seg_flat[r]*C2 + c];
  f2[idx] = (f16)v;
}

// ==================================================================
// Tiled MLP GEMM: block = 256 thr (8 waves), tile M=64 x N=128.
// A tile double-buffered in LDS: async-stage tile k+1 while WMMAs
// consume tile k; one s_wait_asynccnt + one barrier per iteration.
// Each wave owns a 16-col strip and 4 row subtiles => 4 WMMAs/kstep.
// ==================================================================
__global__ void k_mlp1_wmma(const f16* __restrict__ x, const f16* __restrict__ w1t,
                            const float* __restrict__ b1, f16* __restrict__ hbuf) {
  __shared__ __align__(16) f16 As[2*64*32];
  int rt = blockIdx.x;                 // 0..1567  (rows rt*64..)
  int ng = blockIdx.y;                 // 0..3     (cols ng*128..)
  int tid = threadIdx.x;
  int wave = tid >> 5, lane = tid & 31;
  int half = lane >> 4, lm = lane & 15;
  int col = ng*128 + wave*16 + lm;     // this lane's B column
  const f16* brow = w1t + (size_t)col*C2;
  int soff = (tid >> 2)*32 + (tid & 3)*8;       // staging: 1x16B per thread
  const f16* sg = x + (size_t)(rt*64 + (tid >> 2))*C2 + (tid & 3)*8;
  v8f acc[4] = {};
  stage16(sg, As + soff);              // prologue: stage tile 0
  stage_join();
  __syncthreads();
  #pragma unroll 2
  for (int ks = 0; ks < 24; ++ks) {
    int kb = ks*32;
    if (ks + 1 < 24)                   // async-stage next tile into alternate buffer
      stage16(sg + (ks+1)*32, As + ((ks+1)&1)*2048 + soff);
    __builtin_prefetch(brow + kb + 32, 0, 3);
    v8h c0 = *(const v8h*)(brow + kb + half*8);
    v8h c1 = *(const v8h*)(brow + kb + 16 + half*8);
    v16h bf = frag_from_chunks(c0, c1);
    const f16* cur = As + (ks & 1)*2048;
    #pragma unroll
    for (int mt = 0; mt < 4; ++mt) {
      const f16* arow = cur + (mt*16 + lm)*32;
      v8h lo = *(const v8h*)(arow + half*8);
      v8h hi = *(const v8h*)(arow + 16 + half*8);
      v16h a = frag_from_chunks(lo, hi);
      acc[mt] = __builtin_amdgcn_wmma_f32_16x16x32_f16(false, a, false, bf, (short)0, acc[mt], false, false);
    }
    stage_join();                      // next tile landed
    __syncthreads();
  }
  float bc = b1[col];
  #pragma unroll
  for (int mt = 0; mt < 4; ++mt) {
    #pragma unroll
    for (int g = 0; g < 8; ++g) {
      int row = rt*64 + mt*16 + g + 8*half;
      float v = acc[mt][g] + bc;
      v = 0.5f * v * (1.0f + erff(v * 0.70710678118654752f));   // exact GELU
      hbuf[(size_t)row*HIDN + col] = (f16)v;
    }
  }
}

__global__ void k_mlp2_wmma(const f16* __restrict__ hbuf, const f16* __restrict__ w2t,
                            const float* __restrict__ b2, const int* __restrict__ seg_flat,
                            const int* __restrict__ winner, float* __restrict__ out) {
  __shared__ __align__(16) f16 As[2*64*32];
  int rt = blockIdx.x;                 // 0..1567
  int ng = blockIdx.y;                 // 0..5
  int tid = threadIdx.x;
  int wave = tid >> 5, lane = tid & 31;
  int half = lane >> 4, lm = lane & 15;
  int col = ng*128 + wave*16 + lm;
  const f16* brow = w2t + (size_t)col*HIDN;
  int soff = (tid >> 2)*32 + (tid & 3)*8;
  const f16* sg = hbuf + (size_t)(rt*64 + (tid >> 2))*HIDN + (tid & 3)*8;
  v8f acc[4] = {};
  stage16(sg, As + soff);
  stage_join();
  __syncthreads();
  #pragma unroll 2
  for (int ks = 0; ks < 16; ++ks) {
    int kb = ks*32;
    if (ks + 1 < 16)
      stage16(sg + (ks+1)*32, As + ((ks+1)&1)*2048 + soff);
    __builtin_prefetch(brow + kb + 32, 0, 3);
    v8h c0 = *(const v8h*)(brow + kb + half*8);
    v8h c1 = *(const v8h*)(brow + kb + 16 + half*8);
    v16h bf = frag_from_chunks(c0, c1);
    const f16* cur = As + (ks & 1)*2048;
    #pragma unroll
    for (int mt = 0; mt < 4; ++mt) {
      const f16* arow = cur + (mt*16 + lm)*32;
      v8h lo = *(const v8h*)(arow + half*8);
      v8h hi = *(const v8h*)(arow + 16 + half*8);
      v16h a = frag_from_chunks(lo, hi);
      acc[mt] = __builtin_amdgcn_wmma_f32_16x16x32_f16(false, a, false, bf, (short)0, acc[mt], false, false);
    }
    stage_join();
    __syncthreads();
  }
  float bc = b2[col];
  #pragma unroll
  for (int mt = 0; mt < 4; ++mt) {
    #pragma unroll
    for (int g = 0; g < 8; ++g) {
      int r = rt*64 + mt*16 + g + 8*half;
      int gid = (r / SP) * NSEG + seg_flat[r];
      if (winner[gid] == r)                                 // last-write-wins scatter
        out[(size_t)gid*C2 + col] = acc[mt][g] + bc;
    }
  }
}

// ---------------------------------------------------------------- host
extern "C" void kernel_launch(void* const* d_in, const int* in_sizes, int n_in,
                              void* d_out, int out_size, void* d_ws, size_t ws_size,
                              hipStream_t stream) {
  const float* img     = (const float*)d_in[0];
  const int*   seg     = (const int*)  d_in[1];
  const float* conv1_w = (const float*)d_in[2];
  const float* conv1_b = (const float*)d_in[3];
  const float* bn1_g   = (const float*)d_in[4];
  const float* bn1_b   = (const float*)d_in[5];
  const float* conv2_w = (const float*)d_in[6];
  const float* conv2_b = (const float*)d_in[7];
  const float* bn2_g   = (const float*)d_in[8];
  const float* bn2_b   = (const float*)d_in[9];
  const float* pos     = (const float*)d_in[10];
  const float* W1      = (const float*)d_in[11];
  const float* b1      = (const float*)d_in[12];
  const float* W2      = (const float*)d_in[13];
  const float* b2      = (const float*)d_in[14];
  float* out = (float*)d_out;

  char* ws = (char*)d_ws;
  size_t off = 0;
  auto take = [&](size_t bytes) -> void* {
    void* p = ws + off;
    off = (off + bytes + 255) & ~(size_t)255;
    return p;
  };
  float* conv1_raw = (float*)take(sizeof(float)*(size_t)BATCH*C1*SP);   // 25.7 MB
  f16*   f1        = (f16*)  take(sizeof(f16)  *(size_t)BATCH*C1*SP);   // 12.8 MB
  f16*   f2        = (f16*)  take(sizeof(f16)  *(size_t)BATCH*C2*SP);   // 154 MB
  f16*   hbuf      = (f16*)  take(sizeof(f16)  *(size_t)ROWS*HIDN);     // 103 MB
  f16*   wA        = (f16*)  take(sizeof(f16)  *(size_t)C2*KC2);
  f16*   w1t       = (f16*)  take(sizeof(f16)  *(size_t)HIDN*C2);
  f16*   w2t       = (f16*)  take(sizeof(f16)  *(size_t)C2*HIDN);
  float* stats1    = (float*)take(sizeof(float)*2*C1);
  float* stats2    = (float*)take(sizeof(float)*2*C2);
  int*   seg_flat  = (int*)  take(sizeof(int)  *ROWS);
  int*   winner    = (int*)  take(sizeof(int)  *BATCH*NSEG);

  // init (deterministic each call)
  k_zero_f32<<<2048, 256, 0, stream>>>(out, BATCH*NSEG*C2);
  k_zero_f32<<<1,    256, 0, stream>>>(stats1, 2*C1);
  k_zero_f32<<<8,    256, 0, stream>>>(stats2, 2*C2);
  k_fill_i32<<<8,    256, 0, stream>>>(winner, BATCH*NSEG, -1);

  // stage 1: conv1 -> stats -> BN+ReLU (f16)
  {
    int total = BATCH*C1*SP;
    k_conv1<<<(total+255)/256, 256, 0, stream>>>(img, conv1_w, conv1_b, conv1_raw);
    k_stats_f32<<<dim3(C1, 64), 256, 0, stream>>>(conv1_raw, stats1, C1);
    k_bn_relu_1<<<(total+255)/256, 256, 0, stream>>>(conv1_raw, stats1, bn1_g, bn1_b, f1);
  }

  // weight prep (f32 -> f16, conv2 re-layout + W1/W2 transpose)
  k_prep_weights<<<(C2*KC2+255)/256, 256, 0, stream>>>(conv2_w, W1, W2, wA, w1t, w2t);

  // stage 2: conv2 (WMMA implicit GEMM, 4 oc-tiles/wave) -> stats -> BN+ReLU in place
  k_conv2_wmma<<<dim3(C2/(16*OCPW), BATCH*HF*(HF/16)), 32, 0, stream>>>(f1, wA, conv2_b, f2);
  k_stats_f16<<<dim3(C2, 64), 256, 0, stream>>>(f2, stats2, C2);
  {
    int total = BATCH*C2*SP;
    k_bn_relu_2<<<(total+255)/256, 256, 0, stream>>>(f2, stats2, bn2_g, bn2_b);
  }

  // segments: downsample + winner map; then fold pos-embed into f2 (raw-reshape view)
  k_seg<<<(ROWS+255)/256, 256, 0, stream>>>(seg, seg_flat, winner);
  k_addpos<<<(ROWS*C2+255)/256, 256, 0, stream>>>(f2, pos, seg_flat);

  // MLP (tiled WMMA, double-buffered async LDS staging) + fused winner scatter
  k_mlp1_wmma<<<dim3(ROWS/64, HIDN/128), 256, 0, stream>>>(f2, w1t, b1, hbuf);
  k_mlp2_wmma<<<dim3(ROWS/64, C2/128),   256, 0, stream>>>(hbuf, w2t, b2, seg_flat, winner, out);
}
